// Basic_RSNN_ALIF_88845693485079
// MI455X (gfx1250) — matrix operand order repaired and checked by reading, not compiled, not run on hardware
//
#include <hip/hip_runtime.h>
#include <hip/hip_bf16.h>

// ---------------------------------------------------------------------------
// RSNN-ALIF on gfx1250: bf16 WMMA GEMMs (f32 accumulate), f32 neuron state.
//  K1: pack W1 / (W_rec masked)^T / W_out into WMMA B-fragment lane layout
//  K2: xproj[t][b][h] = x[b,t,:] @ W1^T       (bf16 WMMA, LDS-staged A)
//  K3: persistent 16-WG recurrence, state in VGPRs, global atomic step
//      barrier, double-buffered GLOBAL_LOAD_ASYNC_TO_LDS_B128 spike staging
//  K4: out[b,t,o]     = spikes[t,b,:] @ W_out^T (bf16 WMMA, N padded 20->32)
// ---------------------------------------------------------------------------

typedef __attribute__((ext_vector_type(16))) __bf16 v16bf;
typedef __attribute__((ext_vector_type(8)))  float  v8f;

#define BB      128
#define TT      500
#define NIN     700
#define HH      1024
#define NOUT    20
#define KB1     22        // ceil(700/32) K-blocks for W1 GEMM
#define KBR     32        // 1024/32      K-blocks for recurrent / output GEMM
#define NWG_REC 16

__device__ __forceinline__ unsigned short f2bf(float f) {
  union { float f; unsigned int u; } c; c.f = f;
  unsigned int u = c.u;
  u += 0x7FFFu + ((u >> 16) & 1u);          // round-to-nearest-even
  return (unsigned short)(u >> 16);
}
__device__ __forceinline__ float bf2f(unsigned short h) {
  union { unsigned int u; float f; } c; c.u = ((unsigned int)h) << 16;
  return c.f;
}

union AFrag { unsigned int u[8]; v16bf v; };
union BFrag { uint4 q[2];        v16bf v; };
union CFrag { v8f v;             float f[8]; };

__device__ __forceinline__ v8f vzero() {
  v8f z;
#pragma unroll
  for (int i = 0; i < 8; ++i) z[i] = 0.f;
  return z;
}

// Async global -> LDS copy (ASYNCcnt-tracked). lds_off = wave LDS byte offset
// (low 32 bits of the generic pointer to a __shared__ object).
// NOTE: no "memory" clobber on the issue -- ordering vs LDS reads is enforced
// by the s_wait_asynccnt (which does clobber memory) + __syncthreads.
// Volatile asms stay ordered among themselves, so issues never pass the wait.
__device__ __forceinline__ void async_load_b128(unsigned lds_off, const void* gaddr) {
  asm volatile("global_load_async_to_lds_b128 %0, %1, off"
               :: "v"(lds_off), "v"(gaddr));
}
__device__ __forceinline__ void wait_async_le4() {
  asm volatile("s_wait_asynccnt 4" ::: "memory");
}
__device__ __forceinline__ void wait_async_0() {
  asm volatile("s_wait_asynccnt 0" ::: "memory");
}
__device__ __forceinline__ unsigned lds_off_of(const void* p) {
  return (unsigned)(unsigned long long)p;   // LDS aperture: low 32 bits = LDS addr
}

// ---------------------------------------------------------------------------
// Pack B-operand: dst fragment layout [ntile][kblk][lane(32)][e(16)] bf16,
// where for lane L: n = ntile*16 + L%16, k = kblk*32 + (L/16)*8 + e + (e>=8 ? 8 : 0).
// src is row-major [n][k] with leading dim ld; zero-pad OOB; optional diag mask.
// ---------------------------------------------------------------------------
__global__ void pack_b(const float* __restrict__ src, unsigned short* __restrict__ dst,
                       int ld, int nmax, int kmax, int kblks, int maskdiag) {
  const int bx   = blockIdx.x;          // = ntile*kblks + kb
  const int kb   = bx % kblks;
  const int ntl  = bx / kblks;
  const int lane = threadIdx.x;
  const int n    = ntl * 16 + (lane & 15);
  const int kb0  = kb * 32 + (lane >> 4) * 8;
  unsigned short* d = dst + ((size_t)bx * 32 + lane) * 16;
#pragma unroll
  for (int e = 0; e < 16; ++e) {
    int k = kb0 + e + (e >= 8 ? 8 : 0);
    float v = 0.f;
    if (k < kmax && n < nmax && !(maskdiag && k == n)) v = src[(size_t)n * ld + k];
    d[e] = f2bf(v);
  }
}

__global__ void init_counter(unsigned int* c) { *c = 0u; }

// ---------------------------------------------------------------------------
// K2: xproj GEMM. grid = (T, 8). Each WG: all 128 batch rows x 128 H columns.
// 8 waves -> wave (wm,wn): 32-row x 64-col patch = 2x4 WMMA tiles.
// ---------------------------------------------------------------------------
__global__ void __launch_bounds__(256) xproj_gemm(const float* __restrict__ x,
                                                  const unsigned short* __restrict__ pW1,
                                                  unsigned short* __restrict__ xproj) {
  __shared__ unsigned short As[128 * 40];     // 32-K tile, padded stride 40
  const int t    = blockIdx.x;
  const int hb   = blockIdx.y * 128;
  const int tid  = threadIdx.x;
  const int wave = tid >> 5, lane = tid & 31;
  const int lanen = lane & 15, half = lane >> 4;
  const int m0 = (wave & 3) * 32;
  const int n0 = (wave >> 2) * 64;

  v8f acc[2][4];
#pragma unroll
  for (int i = 0; i < 2; ++i)
#pragma unroll
    for (int j = 0; j < 4; ++j) acc[i][j] = vzero();

  const int rr = tid >> 1, hh2 = tid & 1;     // stage: row rr, k-half hh2
  const float* srcrow = x + (size_t)rr * ((size_t)TT * NIN) + (size_t)t * NIN;

  for (int kb = 0; kb < KB1; ++kb) {
    __syncthreads();
    {
      const int kbase = kb * 32 + hh2 * 16;
#pragma unroll
      for (int j = 0; j < 16; ++j) {
        int k = kbase + j;
        float v = (k < NIN) ? srcrow[k] : 0.f;
        As[rr * 40 + hh2 * 16 + j] = f2bf(v);
      }
    }
    if (kb + 1 < KB1)   // prefetch next packed-W1 block (global_prefetch path)
      __builtin_prefetch(pW1 + ((size_t)(((hb + n0) >> 4) * KB1 + kb + 1) * 32 + lane) * 16, 0, 1);
    __syncthreads();

    AFrag a[2];
#pragma unroll
    for (int mt = 0; mt < 2; ++mt) {
      const unsigned short* row = &As[(m0 + mt * 16 + lanen) * 40];
      const int k0 = half * 8, k1 = 16 + half * 8;
#pragma unroll
      for (int j = 0; j < 4; ++j) a[mt].u[j]     = *(const unsigned int*)&row[k0 + 2 * j];
#pragma unroll
      for (int j = 0; j < 4; ++j) a[mt].u[4 + j] = *(const unsigned int*)&row[k1 + 2 * j];
    }
#pragma unroll
    for (int nt = 0; nt < 4; ++nt) {
      const int ntile = (hb + n0 + nt * 16) >> 4;
      const uint4* bp = (const uint4*)(pW1 + ((size_t)(ntile * KB1 + kb) * 32 + lane) * 16);
      BFrag b; b.q[0] = bp[0]; b.q[1] = bp[1];
#pragma unroll
      for (int mt = 0; mt < 2; ++mt)
        acc[mt][nt] = __builtin_amdgcn_wmma_f32_16x16x32_bf16(false, a[mt].v, false, b.v,
                                                              (short)0, acc[mt][nt], false, false);
    }
  }

  unsigned short* dst = xproj + (size_t)t * BB * HH;
#pragma unroll
  for (int mt = 0; mt < 2; ++mt)
#pragma unroll
    for (int nt = 0; nt < 4; ++nt) {
      CFrag c; c.v = acc[mt][nt];
      const int h = hb + n0 + nt * 16 + lanen;
#pragma unroll
      for (int r = 0; r < 8; ++r) {
        const int b = m0 + mt * 16 + half * 8 + r;
        dst[(size_t)b * HH + h] = f2bf(c.f[r]);
      }
    }
}

// ---------------------------------------------------------------------------
// K3: persistent recurrence. 16 WGs; WG wg owns H columns [wg*64, wg*64+64).
// mem/b/spk state held in registers for all 500 steps. Spike A-tiles staged
// via double-buffered async global->LDS copies overlapped with WMMA.
// ---------------------------------------------------------------------------
__global__ void __launch_bounds__(256) rsnn_recurrent(const unsigned short* __restrict__ xproj,
                                                      const unsigned short* __restrict__ pWr,
                                                      unsigned short* __restrict__ spikes,
                                                      const float* __restrict__ tau_h,
                                                      const float* __restrict__ tau_o,
                                                      unsigned int* __restrict__ counter) {
  __shared__ unsigned short As[2][128 * 72];  // double-buffered 64-K tile, stride 72
  const int wg    = blockIdx.x;
  const int nbase = wg * 64;
  const int tid   = threadIdx.x;
  const int wave  = tid >> 5, lane = tid & 31;
  const int lanen = lane & 15, half = lane >> 4;
  const int m0 = (wave & 3) * 32;
  const int n0 = (wave >> 2) * 32;

  float th[2], to2[2];
#pragma unroll
  for (int nt = 0; nt < 2; ++nt) {
    const int h = nbase + n0 + nt * 16 + lanen;
    th[nt]  = tau_h[h];
    to2[nt] = tau_o[h];
  }

  float memv[2][2][8], bad[2][2][8], spv[2][2][8];
#pragma unroll
  for (int mt = 0; mt < 2; ++mt)
#pragma unroll
    for (int nt = 0; nt < 2; ++nt)
#pragma unroll
      for (int r = 0; r < 8; ++r) { memv[mt][nt][r] = 0.f; bad[mt][nt][r] = 0.01f; spv[mt][nt][r] = 0.f; }

  // staging role: row rr (0..127), half hh2 -> 64 bytes = 4 x b128 per thread
  const int rr = tid >> 1, hh2 = tid & 1;
  const unsigned ldsb[2] = { lds_off_of(&As[0][rr * 72 + hh2 * 32]),
                             lds_off_of(&As[1][rr * 72 + hh2 * 32]) };

  for (int t = 0; t < TT; ++t) {
    v8f acc[2][2];
#pragma unroll
    for (int i = 0; i < 2; ++i)
#pragma unroll
      for (int j = 0; j < 2; ++j) acc[i][j] = vzero();

    if (t > 0) {                                          // spk(t=0 input) == 0
      const char* spb = (const char*)(spikes + (size_t)(t - 1) * BB * HH)
                        + (size_t)rr * (HH * 2) + hh2 * 64;
      { // prologue: issue block 0 -> buffer 0
#pragma unroll
        for (int j = 0; j < 4; ++j) async_load_b128(ldsb[0] + j * 16, spb + j * 16);
      }
      for (int blk = 0; blk < 16; ++blk) {
        const int cur = blk & 1;
        if (blk + 1 < 16) {   // issue next block into other buffer, keep it in flight
          const char* g = spb + (blk + 1) * 128;
#pragma unroll
          for (int j = 0; j < 4; ++j) async_load_b128(ldsb[cur ^ 1] + j * 16, g + j * 16);
          wait_async_le4();   // 8 in flight -> <=4 : block `blk` retired (in-order)
        } else {
          wait_async_0();
        }
        __syncthreads();      // all waves' async data for As[cur] landed

#pragma unroll
        for (int ks = 0; ks < 64; ks += 32) {
          AFrag a[2];
#pragma unroll
          for (int mt = 0; mt < 2; ++mt) {
            const unsigned short* row = &As[cur][(m0 + mt * 16 + lanen) * 72];
            const int k0 = ks + half * 8, k1 = ks + 16 + half * 8;
#pragma unroll
            for (int j = 0; j < 4; ++j) a[mt].u[j]     = *(const unsigned int*)&row[k0 + 2 * j];
#pragma unroll
            for (int j = 0; j < 4; ++j) a[mt].u[4 + j] = *(const unsigned int*)&row[k1 + 2 * j];
          }
          const int kblk = (blk * 64 + ks) >> 5;
#pragma unroll
          for (int nt = 0; nt < 2; ++nt) {
            const int ntile = (nbase + n0 + nt * 16) >> 4;
            const uint4* bp = (const uint4*)(pWr + ((size_t)(ntile * KBR + kblk) * 32 + lane) * 16);
            BFrag b; b.q[0] = bp[0]; b.q[1] = bp[1];
#pragma unroll
            for (int mt = 0; mt < 2; ++mt)
              acc[mt][nt] = __builtin_amdgcn_wmma_f32_16x16x32_bf16(false, a[mt].v, false, b.v,
                                                                    (short)0, acc[mt][nt], false, false);
          }
        }
        __syncthreads();      // buffer `cur` free for reuse at blk+2's issue
      }
    }

    // ALIF neuron update (f32, state in registers)
    const unsigned short* xpt = xproj  + (size_t)t * BB * HH;
    unsigned short*       spt = spikes + (size_t)t * BB * HH;
#pragma unroll
    for (int mt = 0; mt < 2; ++mt)
#pragma unroll
      for (int nt = 0; nt < 2; ++nt) {
        CFrag c; c.v = acc[mt][nt];
        const int h = nbase + n0 + nt * 16 + lanen;
#pragma unroll
        for (int r = 0; r < 8; ++r) {
          const int b = m0 + mt * 16 + half * 8 + r;
          const size_t idx = (size_t)b * HH + h;
          const float I     = bf2f(xpt[idx]) + c.f[r];
          const float bn    = to2[nt] * bad[mt][nt][r] + (1.f - to2[nt]) * spv[mt][nt][r];
          const float theta = 0.01f + 1.8f * bn;
          const float mv    = th[nt] * memv[mt][nt][r] + (1.f - th[nt]) * I - theta * spv[mt][nt][r];
          const float s     = (mv - theta) > 0.f ? 1.f : 0.f;
          memv[mt][nt][r] = mv; bad[mt][nt][r] = bn; spv[mt][nt][r] = s;
          spt[idx] = f2bf(s);                     // 0/1 exact in bf16
        }
      }

    // device-wide step barrier
    __threadfence();
    __syncthreads();
    if (tid == 0) {
      __hip_atomic_fetch_add(counter, 1u, __ATOMIC_RELEASE, __HIP_MEMORY_SCOPE_AGENT);
      const unsigned int target = (unsigned int)(t + 1) * NWG_REC;
      while (__hip_atomic_load(counter, __ATOMIC_ACQUIRE, __HIP_MEMORY_SCOPE_AGENT) < target)
        __builtin_amdgcn_s_sleep(1);
    }
    __syncthreads();
    __threadfence();
  }
}

// ---------------------------------------------------------------------------
// K4: output GEMM. Rows = t*B+b of spikes; N padded to 32 (2 WMMA N-tiles),
// cols >= 20 zero-packed. Each wave owns one 16-row tile; grid = 500 WGs.
// ---------------------------------------------------------------------------
__global__ void __launch_bounds__(256) out_gemm(const unsigned short* __restrict__ spikes,
                                                const unsigned short* __restrict__ pWo,
                                                float* __restrict__ out) {
  const int tid   = threadIdx.x;
  const int wave  = tid >> 5, lane = tid & 31;
  const int lanen = lane & 15, half = lane >> 4;
  const int mtile = blockIdx.x * 8 + wave;
  const int m0    = mtile * 16;
  const unsigned short* arow = spikes + (size_t)(m0 + lanen) * HH;

  v8f acc0 = vzero(), acc1 = vzero();
  for (int kb = 0; kb < KBR; ++kb) {
    AFrag a;
    const int k0 = kb * 32 + half * 8, k1 = kb * 32 + 16 + half * 8;
#pragma unroll
    for (int j = 0; j < 4; ++j) a.u[j]     = *(const unsigned int*)&arow[k0 + 2 * j];
#pragma unroll
    for (int j = 0; j < 4; ++j) a.u[4 + j] = *(const unsigned int*)&arow[k1 + 2 * j];
    const uint4* b0p = (const uint4*)(pWo + ((size_t)(0 * KBR + kb) * 32 + lane) * 16);
    const uint4* b1p = (const uint4*)(pWo + ((size_t)(1 * KBR + kb) * 32 + lane) * 16);
    BFrag b0, b1;
    b0.q[0] = b0p[0]; b0.q[1] = b0p[1];
    b1.q[0] = b1p[0]; b1.q[1] = b1p[1];
    acc0 = __builtin_amdgcn_wmma_f32_16x16x32_bf16(false, a.v, false, b0.v, (short)0, acc0, false, false);
    acc1 = __builtin_amdgcn_wmma_f32_16x16x32_bf16(false, a.v, false, b1.v, (short)0, acc1, false, false);
  }

  CFrag c0, c1; c0.v = acc0; c1.v = acc1;
#pragma unroll
  for (int r = 0; r < 8; ++r) {
    const int m  = m0 + half * 8 + r;     // = t*B + b
    const int tt = m >> 7;
    const int b  = m & 127;
    float* orow = out + ((size_t)b * TT + tt) * NOUT;
    orow[lanen] = c0.f[r];
    if (lanen < 4) orow[16 + lanen] = c1.f[r];
  }
}

// ---------------------------------------------------------------------------
extern "C" void kernel_launch(void* const* d_in, const int* in_sizes, int n_in,
                              void* d_out, int out_size, void* d_ws, size_t ws_size,
                              hipStream_t stream) {
  (void)in_sizes; (void)n_in; (void)out_size; (void)ws_size;
  const float* x    = (const float*)d_in[0];
  const float* W1   = (const float*)d_in[1];
  const float* Wrec = (const float*)d_in[2];
  const float* Wout = (const float*)d_in[3];
  const float* tauh = (const float*)d_in[4];
  const float* tauo = (const float*)d_in[5];

  char* ws = (char*)d_ws;
  const size_t szTile = (size_t)TT * BB * HH * 2;          // 131,072,000 B
  const size_t szW1   = (size_t)64 * KB1 * 32 * 16 * 2;    // 1,441,792 B
  const size_t szWr   = (size_t)64 * KBR * 32 * 16 * 2;    // 2,097,152 B
  const size_t szWo   = (size_t)2  * KBR * 32 * 16 * 2;    // 65,536 B

  unsigned short* xproj   = (unsigned short*)(ws);
  unsigned short* spikes  = (unsigned short*)(ws + szTile);
  unsigned short* pW1     = (unsigned short*)(ws + 2 * szTile);
  unsigned short* pWr     = (unsigned short*)(ws + 2 * szTile + szW1);
  unsigned short* pWo     = (unsigned short*)(ws + 2 * szTile + szW1 + szWr);
  unsigned int*   counter = (unsigned int*)  (ws + 2 * szTile + szW1 + szWr + szWo);

  init_counter<<<1, 1, 0, stream>>>(counter);
  pack_b<<<64 * KB1, 32, 0, stream>>>(W1,   pW1, NIN, HH,   NIN, KB1, 0);
  pack_b<<<64 * KBR, 32, 0, stream>>>(Wrec, pWr, HH,  HH,   HH,  KBR, 1);
  pack_b<<<2  * KBR, 32, 0, stream>>>(Wout, pWo, HH,  NOUT, HH,  KBR, 0);
  xproj_gemm<<<dim3(TT, 8), 256, 0, stream>>>(x, pW1, xproj);
  rsnn_recurrent<<<NWG_REC, 256, 0, stream>>>(xproj, pWr, spikes, tauh, tauo, counter);
  out_gemm<<<(TT * BB) / 128, 256, 0, stream>>>(spikes, pWo, (float*)d_out);
}
